// BaselineModel_58411555225744
// MI455X (gfx1250) — matrix operand order repaired and checked by reading, not compiled, not run on hardware
//
#include <hip/hip_runtime.h>
#include <math.h>
#include <stdint.h>

typedef __bf16 bf16;
typedef __bf16 v8bf  __attribute__((ext_vector_type(8)));
typedef __bf16 v16bf __attribute__((ext_vector_type(16)));
typedef float  v8f   __attribute__((ext_vector_type(8)));

#define GLOBAL_AS __attribute__((address_space(1)))

constexpr int B_  = 32;
constexpr int T_  = 512;
constexpr int F_  = 513;
constexpr int H_  = 512;
constexpr int G_  = 1536;   // 3*H
constexpr int KX_ = 544;    // F padded to multiple of 32
constexpr int NWG = 16;     // persistent workgroups (one h-slice of 32 dims each)
constexpr int NLPAD = 528;  // F padded to multiple of 16 for output linears

// ---- workspace layout (all region sizes are multiples of 256 bytes) ----
constexpr size_t OFF_XBF  = 0;
constexpr size_t SZ_XBF   = (size_t)T_ * B_ * KX_ * 2;
constexpr size_t OFF_WIH1 = OFF_XBF + SZ_XBF;
constexpr size_t SZ_WIH1  = (size_t)G_ * KX_ * 2;
constexpr size_t OFF_WHH1 = OFF_WIH1 + SZ_WIH1;
constexpr size_t SZ_WH    = (size_t)G_ * H_ * 2;
constexpr size_t OFF_WIH2 = OFF_WHH1 + SZ_WH;
constexpr size_t OFF_WHH2 = OFF_WIH2 + SZ_WH;
constexpr size_t OFF_WIH3 = OFF_WHH2 + SZ_WH;
constexpr size_t OFF_WHH3 = OFF_WIH3 + SZ_WH;
constexpr size_t OFF_WL1  = OFF_WHH3 + SZ_WH;
constexpr size_t SZ_WL    = (size_t)NLPAD * H_ * 2;
constexpr size_t OFF_WL2  = OFF_WL1 + SZ_WL;
constexpr size_t OFF_HA   = OFF_WL2 + SZ_WL;
constexpr size_t SZ_H3    = (size_t)3 * B_ * H_ * 2;
constexpr size_t OFF_HB   = OFF_HA + SZ_H3;
constexpr size_t OFF_OUTB = OFF_HB + SZ_H3;
constexpr size_t SZ_OUTB  = (size_t)T_ * B_ * H_ * 2;
constexpr size_t OFF_BAR  = OFF_OUTB + SZ_OUTB;

// ---- force global (addrspace(1)) 16B loads ------------------------------
__device__ __forceinline__ v8bf gload8(const bf16* p) {
  const GLOBAL_AS v8bf* gp = (const GLOBAL_AS v8bf*)(uintptr_t)p;
  return *gp;
}

// ---- WMMA fragment loads (bf16, 16x16x32) -------------------------------
// A operand: lane half h holds K in [16h, 16h+16), contiguous
__device__ __forceinline__ v16bf ldfragA(const bf16* row, int kk, int hf) {
  const bf16* p = row + kk * 32 + hf * 16;
  v8bf lo = gload8(p);
  v8bf hi = gload8(p + 8);
  return __builtin_shufflevector(lo, hi, 0,1,2,3,4,5,6,7,8,9,10,11,12,13,14,15);
}
// B operand: lane half h holds K in [16h, 16h+16), contiguous
__device__ __forceinline__ v16bf ldfragB(const bf16* row, int kk, int hf) {
  const bf16* p = row + kk * 32 + hf * 16;
  v8bf lo = gload8(p);
  v8bf hi = gload8(p + 8);
  return __builtin_shufflevector(lo, hi, 0,1,2,3,4,5,6,7,8,9,10,11,12,13,14,15);
}

__device__ __forceinline__ v8f wmma_bf16f32(v16bf a, v16bf b, v8f c) {
  return __builtin_amdgcn_wmma_f32_16x16x32_bf16(false, a, false, b, (short)0, c,
                                                 false, false);
}

// ---- device-wide sense-reversing barrier (16 co-resident blocks) --------
__device__ __forceinline__ void grid_bar(int* bar, int* gen) {
  __syncthreads();
  if (threadIdx.x == 0) {
    __threadfence();                       // release h writes to device scope
    int g = *gen;
    if (atomicAdd(&bar[0], 1) == NWG - 1) {
      atomicExch(&bar[0], 0);
      __threadfence();
      atomicAdd(&bar[1], 1);
    } else {
      while (atomicAdd(&bar[1], 0) <= g) __builtin_amdgcn_s_sleep(2);
    }
    *gen = g + 1;
  }
  __threadfence();                         // acquire: invalidate stale lines
  __syncthreads();
}

// ---- prep: fp32 -> bf16 padded copies, zero h-state and barrier ---------
__global__ void prep(const float* __restrict__ x,
                     const float* __restrict__ wih1, const float* __restrict__ whh1,
                     const float* __restrict__ wih2, const float* __restrict__ whh2,
                     const float* __restrict__ wih3, const float* __restrict__ whh3,
                     const float* __restrict__ wl1,  const float* __restrict__ wl2,
                     char* __restrict__ ws) {
  const size_t tid = (size_t)blockIdx.x * blockDim.x + threadIdx.x;
  const size_t stp = (size_t)gridDim.x * blockDim.x;

  // x [B,T,F] -> xbf [T,B,KX] (zero padded)
  bf16* xbf = (bf16*)(ws + OFF_XBF);
  for (size_t k = tid; k < (size_t)T_ * B_ * KX_; k += stp) {
    size_t t = k / (B_ * KX_), r = k % (B_ * KX_), b = r / KX_, c = r % KX_;
    xbf[k] = (c < (size_t)F_) ? (bf16)x[(b * T_ + t) * F_ + c] : (bf16)0.f;
  }
  // w_ih1 [G,F] -> [G,KX] padded
  bf16* w1 = (bf16*)(ws + OFF_WIH1);
  for (size_t k = tid; k < (size_t)G_ * KX_; k += stp) {
    size_t n = k / KX_, c = k % KX_;
    w1[k] = (c < (size_t)F_) ? (bf16)wih1[n * F_ + c] : (bf16)0.f;
  }
  // direct converts [G,H]
  {
    const float* src[5] = {whh1, wih2, whh2, wih3, whh3};
    const size_t off[5] = {OFF_WHH1, OFF_WIH2, OFF_WHH2, OFF_WIH3, OFF_WHH3};
    for (int i = 0; i < 5; i++) {
      bf16* d = (bf16*)(ws + off[i]);
      const float* s = src[i];
      for (size_t k = tid; k < (size_t)G_ * H_; k += stp) d[k] = (bf16)s[k];
    }
  }
  // w_l [F,H] -> [528,H] padded rows
  {
    const float* src[2] = {wl1, wl2};
    const size_t off[2] = {OFF_WL1, OFF_WL2};
    for (int i = 0; i < 2; i++) {
      bf16* d = (bf16*)(ws + off[i]);
      const float* s = src[i];
      for (size_t k = tid; k < (size_t)NLPAD * H_; k += stp) {
        size_t n = k / H_, c = k % H_;
        d[k] = (n < (size_t)F_) ? (bf16)s[n * H_ + c] : (bf16)0.f;
      }
    }
  }
  // h0 = 0 (both ping-pong buffers), barrier = 0
  bf16* ha = (bf16*)(ws + OFF_HA);
  bf16* hb = (bf16*)(ws + OFF_HB);
  for (size_t k = tid; k < (size_t)3 * B_ * H_; k += stp) {
    ha[k] = (bf16)0.f; hb[k] = (bf16)0.f;
  }
  if (tid < 16) ((int*)(ws + OFF_BAR))[tid] = 0;
}

// ---- persistent sequential GRU: 16 blocks x 256 threads (8 wave32) ------
__global__ __launch_bounds__(256, 1) void gru_seq(
    const float* __restrict__ bih1, const float* __restrict__ bhh1,
    const float* __restrict__ bih2, const float* __restrict__ bhh2,
    const float* __restrict__ bih3, const float* __restrict__ bhh3,
    char* __restrict__ ws) {
  const bf16* xbf = (const bf16*)(ws + OFF_XBF);
  const bf16* WIH[3] = {(const bf16*)(ws + OFF_WIH1), (const bf16*)(ws + OFF_WIH2),
                        (const bf16*)(ws + OFF_WIH3)};
  const bf16* WHH[3] = {(const bf16*)(ws + OFF_WHH1), (const bf16*)(ws + OFF_WHH2),
                        (const bf16*)(ws + OFF_WHH3)};
  const float* BIH[3] = {bih1, bih2, bih3};
  const float* BHH[3] = {bhh1, bhh2, bhh3};
  bf16* hOld[3];
  bf16* hNew[3];
  {
    bf16* ha = (bf16*)(ws + OFF_HA);
    bf16* hb = (bf16*)(ws + OFF_HB);
    for (int l = 0; l < 3; l++) { hOld[l] = ha + l * B_ * H_; hNew[l] = hb + l * B_ * H_; }
  }
  bf16* outb = (bf16*)(ws + OFF_OUTB);
  int* bar = (int*)(ws + OFF_BAR);

  const int wg   = blockIdx.x;          // h-dim slice [32*wg, 32*wg+32)
  const int lane = threadIdx.x & 31;
  // wave-uniform scalar: keep tile decode + loop control in SGPRs
  const int wave = __builtin_amdgcn_readfirstlane(threadIdx.x >> 5);
  const int hf   = lane >> 4;
  const int l16  = lane & 15;
  int gen = 0;

  __shared__ float gi_s[32][100];  // gates from input GEMM  (cols: r|z|n of 32 dims)
  __shared__ float gh_s[32][100];  // gates from hidden GEMM

  for (int t = 0; t < T_; t++) {
#pragma unroll
    for (int l = 0; l < 3; l++) {
      const bf16* Agi  = (l == 0) ? (xbf + (size_t)t * B_ * KX_) : hNew[l - 1];
      const int ldagi  = (l == 0) ? KX_ : H_;
      const bf16* Agh  = hOld[l];
      const bf16* Wgi  = WIH[l];
      const bf16* Wgh  = WHH[l];

      // 24 output tiles per WG: {gi,gh} x 2 m-tiles x 6 n-tiles; 3 per wave
#pragma unroll
      for (int i = 0; i < 3; i++) {
        const int tt = wave * 3 + i;                       // scalar
        const int mat = tt / 12, rm = tt % 12, mt = rm / 6, nt = rm % 6;
        const int g = nt >> 1, ch = nt & 1;
        const bf16* A = mat ? Agh : Agi;
        const bf16* W = mat ? Wgh : Wgi;
        const int lda = mat ? H_ : ldagi;
        const bf16* ap = A + (size_t)(mt * 16 + l16) * lda;
        const bf16* wp = W + (size_t)(g * 512 + wg * 32 + ch * 16 + l16) * lda;
        v8f c = {0.f, 0.f, 0.f, 0.f, 0.f, 0.f, 0.f, 0.f};
#pragma unroll 4
        for (int kk = 0; kk < 16; kk++) {                  // K = 512
          v16bf af  = ldfragA(ap, kk, hf);
          v16bf bfr = ldfragB(wp, kk, hf);
          c = wmma_bf16f32(af, bfr, c);
        }
        if ((l == 0) && (mat == 0)) {                      // K tail 512..543
          v16bf af  = ldfragA(ap, 16, hf);
          v16bf bfr = ldfragB(wp, 16, hf);
          c = wmma_bf16f32(af, bfr, c);
        }
        float* dst = mat ? &gh_s[0][0] : &gi_s[0][0];
        const int col  = g * 32 + ch * 16 + l16;
        const int mrow = mt * 16 + hf * 8;
#pragma unroll
        for (int v = 0; v < 8; v++) dst[(mrow + v) * 100 + col] = c[v];
      }
      __syncthreads();

      // pointwise GRU update for our 32 dims x 32 batch rows
      const float* bi = BIH[l];
      const float* bh = BHH[l];
      for (int e = threadIdx.x; e < B_ * 32; e += 256) {
        int m = e >> 5, j = e & 31;
        int gc = wg * 32 + j;
        float ir  = gi_s[m][j]      + bi[gc];
        float iz  = gi_s[m][32 + j] + bi[512 + gc];
        float inn = gi_s[m][64 + j] + bi[1024 + gc];
        float hr  = gh_s[m][j]      + bh[gc];
        float hz  = gh_s[m][32 + j] + bh[512 + gc];
        float hn  = gh_s[m][64 + j] + bh[1024 + gc];
        float r = 1.f / (1.f + __expf(-(ir + hr)));
        float z = 1.f / (1.f + __expf(-(iz + hz)));
        float n = tanhf(inn + r * hn);
        float ho = (float)hOld[l][m * H_ + gc];
        float hv = (1.f - z) * n + z * ho;
        hNew[l][m * H_ + gc] = (bf16)hv;
        if (l == 2) outb[((size_t)t * B_ + m) * H_ + gc] = (bf16)hv;
      }
      grid_bar(bar, &gen);  // publish hNew[l] to all WGPs
    }
#pragma unroll
    for (int l = 0; l < 3; l++) { bf16* tmp = hOld[l]; hOld[l] = hNew[l]; hNew[l] = tmp; }
  }
}

// ---- output linears + mask: M=16384, N=528(x2), K=512 -------------------
__global__ __launch_bounds__(256, 2) void mask_out(
    const float* __restrict__ x, const float* __restrict__ bl1,
    const float* __restrict__ bl2, const char* __restrict__ ws,
    float* __restrict__ out) {
  const bf16* A  = (const bf16*)(ws + OFF_OUTB);   // [T*B, H]
  const bf16* W1 = (const bf16*)(ws + OFF_WL1);    // [528, H]
  const bf16* W2 = (const bf16*)(ws + OFF_WL2);
  const int lane = threadIdx.x & 31, hf = lane >> 4, l16 = lane & 15;
  const int wid = __builtin_amdgcn_readfirstlane((blockIdx.x * 256 + threadIdx.x) >> 5);
  const int nNt = NLPAD / 16;                             // 33
  const int mt = wid / nNt, nt = wid % nNt;

  const bf16* ap  = A  + (size_t)(mt * 16 + l16) * H_;
  const bf16* w1p = W1 + (size_t)(nt * 16 + l16) * H_;
  const bf16* w2p = W2 + (size_t)(nt * 16 + l16) * H_;
  v8f c1 = {0.f, 0.f, 0.f, 0.f, 0.f, 0.f, 0.f, 0.f};
  v8f c2 = {0.f, 0.f, 0.f, 0.f, 0.f, 0.f, 0.f, 0.f};
#pragma unroll 4
  for (int kk = 0; kk < H_ / 32; kk++) {
    v16bf af = ldfragA(ap, kk, hf);
    v16bf b1 = ldfragB(w1p, kk, hf);
    v16bf b2 = ldfragB(w2p, kk, hf);
    c1 = wmma_bf16f32(af, b1, c1);
    c2 = wmma_bf16f32(af, b2, c2);
  }
  int f = nt * 16 + l16;
  if (f < F_) {
    float bb1 = bl1[f], bb2 = bl2[f];
#pragma unroll
    for (int v = 0; v < 8; v++) {
      int m = mt * 16 + hf * 8 + v;   // row = t*32 + b
      int tt = m >> 5, b = m & 31;
      size_t o = ((size_t)b * T_ + tt) * F_ + f;
      float s1 = c1[v] + bb1; s1 = s1 > 0.f ? s1 : 0.f;
      float s2 = c2[v] + bb2; s2 = s2 > 0.f ? s2 : 0.f;
      float xv = x[o];
      float d = s1 + s2 + 1e-16f;
      out[o] = (s1 / d) * xv;
      out[(size_t)B_ * T_ * F_ + o] = (s2 / d) * xv;
    }
  }
}

extern "C" void kernel_launch(void* const* d_in, const int* in_sizes, int n_in,
                              void* d_out, int out_size, void* d_ws, size_t ws_size,
                              hipStream_t stream) {
  (void)in_sizes; (void)n_in; (void)out_size; (void)ws_size;
  const float* x    = (const float*)d_in[0];
  const float* wih1 = (const float*)d_in[1];
  const float* whh1 = (const float*)d_in[2];
  const float* bih1 = (const float*)d_in[3];
  const float* bhh1 = (const float*)d_in[4];
  const float* wih2 = (const float*)d_in[5];
  const float* whh2 = (const float*)d_in[6];
  const float* bih2 = (const float*)d_in[7];
  const float* bhh2 = (const float*)d_in[8];
  const float* wih3 = (const float*)d_in[9];
  const float* whh3 = (const float*)d_in[10];
  const float* bih3 = (const float*)d_in[11];
  const float* bhh3 = (const float*)d_in[12];
  const float* wl1  = (const float*)d_in[13];
  const float* bl1  = (const float*)d_in[14];
  const float* wl2  = (const float*)d_in[15];
  const float* bl2  = (const float*)d_in[16];
  char* ws = (char*)d_ws;
  float* out = (float*)d_out;

  prep<<<1024, 256, 0, stream>>>(x, wih1, whh1, wih2, whh2, wih3, whh3, wl1, wl2, ws);
  gru_seq<<<NWG, 256, 0, stream>>>(bih1, bhh1, bih2, bhh2, bih3, bhh3, ws);
  // 1024 m-tiles * 33 n-tiles = 33792 waves = 4224 blocks of 8 waves
  mask_out<<<4224, 256, 0, stream>>>(x, bl1, bl2, ws, out);
}
